// SparseDyadicAttention_55284819034471
// MI455X (gfx1250) — matrix operand order repaired and checked by reading, not compiled, not run on hardware
//
#include <hip/hip_runtime.h>
#include <hip/hip_bf16.h>
#include <math.h>

// ---------------------------------------------------------------------------
// SparseDyadicAttention for MI455X (gfx1250, wave32, WMMA 16x16x32 f16).
//
// Decomposition (all matmuls on v_wmma_f32_16x16x32_f16, f32 accumulate):
//   1. Qp  = x @ M   (f16 out)      2. Val = x @ V (f16 out)
//   3. RoPE(Qp) in place; Kr = RoPE(x) once (head-invariant)
//   4. P   = softmax(causal(Qr Kr^T / sqrt(768)))  (f16, 134 MB -> L2 resident)
//   5. out = sum_h P_h @ Val_h  (f32)
//
// gfx1250 specifics used: v_wmma_f32_16x16x32_f16, global_load_async_to_lds_b128
// (+ s_wait_asynccnt) for pure-copy tile staging, global_prefetch_b8, wave32
// fragment layouts per CDNA5 ISA 7.12.2.
// ---------------------------------------------------------------------------

#define B_   2
#define T_   2048
#define D_   768
#define H_   8
#define HD_  6144   // H_*D_
#define BTM_ 4096   // B_*T_ rows for the projection GEMMs

typedef _Float16 v16h __attribute__((ext_vector_type(16)));
typedef _Float16 v8h  __attribute__((ext_vector_type(8)));
typedef float    v8f  __attribute__((ext_vector_type(8)));

union AFrag { v16h v; v8h h[2]; };

// Async copy of 32 contiguous bytes global -> LDS (two B128 ops, shared
// address registers; inst offset applies to both LDS and memory sides).
__device__ __forceinline__ void async_copy32(const _Float16* gsrc, _Float16* lds) {
  unsigned       laddr = (unsigned)(unsigned long long)lds;   // LDS byte offset
  unsigned long long ga = (unsigned long long)gsrc;
  asm volatile("global_load_async_to_lds_b128 %0, %1, off\n\t"
               "global_load_async_to_lds_b128 %0, %1, off offset:16"
               :: "v"(laddr), "v"(ga) : "memory");
}

__device__ __forceinline__ void async_wait() {
  asm volatile("s_wait_asynccnt 0x0" ::: "memory");
}

// One 64(M)x64(N) x 32(K) tile step: 4 waves, wave w owns rows [16w,16w+16),
// 4 WMMA 16x16 tiles across the 64 columns. Fragment layouts per CDNA5 ISA
// 7.12.2 (wave32):
//   A 16x32 f16 : lanes 0-15 -> M=lane, halves {K0..7, K16..23};
//                 lanes 16-31 -> {K8..15, K24..31}
//   B 32x16 f16 : lane -> col (lane&15), K split same as A
//   C 16x16 f32 : vgpr r -> m = (lane>>4)*8 + r, n = lane&15
__device__ __forceinline__ void mma_tile(const _Float16 (*sA)[32],
                                         const _Float16 (*sB)[32],
                                         v8f acc[4]) {
  const int lane = threadIdx.x & 31;
  const int wv   = threadIdx.x >> 5;
  const int l16  = lane & 15;
  const int lh   = lane >> 4;
  AFrag a;
  a.h[0] = *(const v8h*)&sA[wv * 16 + l16][lh * 8];
  a.h[1] = *(const v8h*)&sA[wv * 16 + l16][lh * 8 + 16];
#pragma unroll
  for (int t = 0; t < 4; ++t) {
    AFrag bfr;
    bfr.h[0] = *(const v8h*)&sB[t * 16 + l16][lh * 8];
    bfr.h[1] = *(const v8h*)&sB[t * 16 + l16][lh * 8 + 16];
    v16h bf = bfr.v;
    acc[t] = __builtin_amdgcn_wmma_f32_16x16x32_f16(
        /*neg_a=*/false, a.v, /*neg_b=*/false, bf,
        /*c_mod=*/(short)0, acc[t], /*reuse_a=*/false, /*reuse_b=*/false);
  }
}

// ---------------------------------------------------------------------------
// Pass 1: C[4096 x 6144](f16) = A[4096 x 768](f32) * W[768 x 6144](f32)
// grid (N/64, M/64), block 128 (4 waves). f32->f16 conversion during staging.
// W tile is transposed through registers: each thread owns one output column,
// reads 16 K-strided f32 (coalesced across lanes), writes one aligned v16h.
// ---------------------------------------------------------------------------
__global__ void proj_gemm_kernel(const float* __restrict__ A,
                                 const float* __restrict__ W,
                                 _Float16* __restrict__ C) {
  __shared__ __align__(64) _Float16 sA[64][32];
  __shared__ __align__(64) _Float16 sB[64][32];  // transposed: [n][k]

  const int nb = blockIdx.x * 64;
  const int mb = blockIdx.y * 64;
  v8f acc[4] = {};

  for (int kb = 0; kb < D_; kb += 32) {
    // Stage A tile 64x32 (f32 -> f16), contiguous per thread
    {
      const int e = threadIdx.x * 16;
      const int m = e >> 5, k0 = e & 31;
      const float* src = A + (size_t)(mb + m) * D_ + kb + k0;
      if (kb + 32 < D_) __builtin_prefetch(src + 32, 0, 1);
      v16h tmp;
#pragma unroll
      for (int i = 0; i < 16; ++i) tmp[i] = (_Float16)src[i];
      *(v16h*)&sA[m][k0] = tmp;
    }
    // Stage W tile 32x64 transposed via registers into sB[n][k]
    {
      const int n  = threadIdx.x & 63;
      const int k0 = (threadIdx.x >> 6) * 16;   // 0 or 16
      const float* src = W + (size_t)(kb + k0) * HD_ + nb + n;
      if (kb + 32 < D_) __builtin_prefetch(src + (size_t)32 * HD_, 0, 1);
      v16h tmp;
#pragma unroll
      for (int i = 0; i < 16; ++i) tmp[i] = (_Float16)src[(size_t)i * HD_];
      *(v16h*)&sB[n][k0] = tmp;
    }
    __syncthreads();
    mma_tile(sA, sB, acc);
    __syncthreads();
  }

  const int lane = threadIdx.x & 31;
  const int wv   = threadIdx.x >> 5;
#pragma unroll
  for (int t = 0; t < 4; ++t) {
    const int n = nb + t * 16 + (lane & 15);
#pragma unroll
    for (int r = 0; r < 8; ++r) {
      const int m = mb + wv * 16 + (lane >> 4) * 8 + r;
      C[(size_t)m * HD_ + n] = (_Float16)acc[t][r];
    }
  }
}

// ---------------------------------------------------------------------------
// Pass 2: RoPE. u in [0,H): rotate Qp head u in place. u==H: build Kr from x.
// One thread per (b,t,u,freq-pair).
// ---------------------------------------------------------------------------
__global__ void rope_kernel(const float* __restrict__ x,
                            const int* __restrict__ positions,
                            _Float16* __restrict__ Qp,
                            _Float16* __restrict__ Kr) {
  const int HALF = D_ / 2;  // 384
  int idx = blockIdx.x * blockDim.x + threadIdx.x;
  const int total = B_ * T_ * (H_ + 1) * HALF;
  if (idx >= total) return;

  const int j = idx % HALF;  idx /= HALF;
  const int u = idx % (H_ + 1); idx /= (H_ + 1);
  const int t = idx % T_;
  const int b = idx / T_;

  const float pos  = (float)positions[t];
  const float invf = __powf(10000.0f, -(float)j * (1.0f / (float)HALF));
  float sn, cs;
  sincosf(pos * invf, &sn, &cs);

  if (u < H_) {
    const size_t base = (size_t)(b * T_ + t) * HD_ + (size_t)u * D_ + j;
    const float a0 = (float)Qp[base];
    const float a1 = (float)Qp[base + HALF];
    Qp[base]        = (_Float16)(a0 * cs - a1 * sn);
    Qp[base + HALF] = (_Float16)(a0 * sn + a1 * cs);
  } else {
    const size_t xb = (size_t)(b * T_ + t) * D_ + j;
    const float a0 = x[xb];
    const float a1 = x[xb + HALF];
    Kr[xb]        = (_Float16)(a0 * cs - a1 * sn);
    Kr[xb + HALF] = (_Float16)(a0 * sn + a1 * cs);
  }
}

// ---------------------------------------------------------------------------
// Pass 3: scores. S(b,h)[q,k] = scale * Qr(b,h,q,:) . Kr(b,k,:)
// grid (T/64 ktiles, T/64 qtiles, B*H). Upper-triangular tiles early-exit;
// softmax only reads j<=q so no in-tile masking needed. Both tiles are pure
// f16 copies -> staged with global_load_async_to_lds_b128 (ASYNCcnt).
// ---------------------------------------------------------------------------
__global__ void scores_kernel(const _Float16* __restrict__ Q,
                              const _Float16* __restrict__ Kr,
                              _Float16* __restrict__ S) {
  if (blockIdx.x > blockIdx.y) return;  // fully masked tile
  __shared__ __align__(64) _Float16 sA[64][32];
  __shared__ __align__(64) _Float16 sB[64][32];

  const int kbT = blockIdx.x * 64;
  const int qbT = blockIdx.y * 64;
  const int bh  = blockIdx.z;
  const int b   = bh >> 3;
  const int h   = bh & 7;
  const float scale = 0.03608439182435161f;  // 1/sqrt(768)

  const _Float16* Abase = Q  + (size_t)(b * T_) * HD_ + (size_t)h * D_;
  const _Float16* Bbase = Kr + (size_t)(b * T_) * D_;
  v8f acc[4] = {};

  const int e  = threadIdx.x * 16;
  const int m  = e >> 5, k0 = e & 31;

  for (int kb = 0; kb < D_; kb += 32) {
    async_copy32(Abase + (size_t)(qbT + m) * HD_ + kb + k0, &sA[m][k0]);
    async_copy32(Bbase + (size_t)(kbT + m) * D_  + kb + k0, &sB[m][k0]);
    async_wait();
    __syncthreads();
    mma_tile(sA, sB, acc);
    __syncthreads();
  }

  const int lane = threadIdx.x & 31;
  const int wv   = threadIdx.x >> 5;
#pragma unroll
  for (int t = 0; t < 4; ++t) {
    const int n = kbT + t * 16 + (lane & 15);
#pragma unroll
    for (int r = 0; r < 8; ++r) {
      const int mq = qbT + wv * 16 + (lane >> 4) * 8 + r;
      S[((size_t)bh * T_ + mq) * T_ + n] = (_Float16)(acc[t][r] * scale);
    }
  }
}

// ---------------------------------------------------------------------------
// Pass 4: row softmax over causal length q+1; zero-fill masked tail so the
// output GEMM can read full tiles. One 256-thread block per row.
// ---------------------------------------------------------------------------
__global__ void softmax_kernel(_Float16* __restrict__ S) {
  __shared__ float red[256];
  const int row = blockIdx.x;
  const int q   = row & (T_ - 1);
  const int valid = q + 1;
  _Float16* Sr = S + (size_t)row * T_;
  const int tid = threadIdx.x;

  float m = -3.0e38f;
  for (int j = tid; j < valid; j += 256) m = fmaxf(m, (float)Sr[j]);
  red[tid] = m;
  __syncthreads();
  for (int s = 128; s > 0; s >>= 1) {
    if (tid < s) red[tid] = fmaxf(red[tid], red[tid + s]);
    __syncthreads();
  }
  m = red[0];
  __syncthreads();

  float sum = 0.0f;
  for (int j = tid; j < valid; j += 256) sum += __expf((float)Sr[j] - m);
  red[tid] = sum;
  __syncthreads();
  for (int s = 128; s > 0; s >>= 1) {
    if (tid < s) red[tid] += red[tid + s];
    __syncthreads();
  }
  const float inv = 1.0f / red[0];

  for (int j = tid; j < valid; j += 256)
    Sr[j] = (_Float16)(__expf((float)Sr[j] - m) * inv);
  for (int j = valid + tid; j < T_; j += 256)
    Sr[j] = (_Float16)0.0f;
}

// ---------------------------------------------------------------------------
// Pass 5: out[b,q,:] = sum_h P(b,h)[q,:] @ Val(b,:,h,:). Head sum folded into
// the K loop (no atomics). K range causally trimmed to the query tile.
// P tile: async copy. Val tile: register transpose (coalesced across lanes).
// grid (768/64 dd-tiles, T/64 q-tiles, B).
// ---------------------------------------------------------------------------
__global__ void out_gemm_kernel(const _Float16* __restrict__ P,
                                const _Float16* __restrict__ Val,
                                float* __restrict__ out) {
  __shared__ __align__(64) _Float16 sA[64][32];
  __shared__ __align__(64) _Float16 sB[64][32];  // [dd][j]

  const int ddb = blockIdx.x * 64;
  const int qb  = blockIdx.y * 64;
  const int b   = blockIdx.z;
  const int jtiles = (qb + 64) / 32;  // causal trim
  v8f acc[4] = {};

  const int e  = threadIdx.x * 16;
  const int m  = e >> 5, k0a = e & 31;
  const int n  = threadIdx.x & 63;
  const int k0 = (threadIdx.x >> 6) * 16;   // 0 or 16

  for (int h = 0; h < H_; ++h) {
    const _Float16* Pb = P   + ((size_t)(b * H_ + h) * T_ + qb) * T_;
    const _Float16* Vb = Val + (size_t)(b * T_) * HD_ + (size_t)h * D_ + ddb;
    for (int jt = 0; jt < jtiles; ++jt) {
      const int jb = jt * 32;
      // P tile 64x32: pure f16 copy -> async to LDS
      async_copy32(Pb + (size_t)m * T_ + jb + k0a, &sA[m][k0a]);
      // Val tile 32x64 transposed via registers into sB[n][k]
      {
        const _Float16* src = Vb + (size_t)(jb + k0) * HD_ + n;
        if (jt + 1 < jtiles) __builtin_prefetch(src + (size_t)32 * HD_, 0, 1);
        v16h tmp;
#pragma unroll
        for (int i = 0; i < 16; ++i) tmp[i] = src[(size_t)i * HD_];
        *(v16h*)&sB[n][k0] = tmp;
      }
      async_wait();
      __syncthreads();
      mma_tile(sA, sB, acc);
      __syncthreads();
    }
  }

  const int lane = threadIdx.x & 31;
  const int wv   = threadIdx.x >> 5;
#pragma unroll
  for (int t = 0; t < 4; ++t) {
    const int nn = ddb + t * 16 + (lane & 15);
#pragma unroll
    for (int r = 0; r < 8; ++r) {
      const int mq = qb + wv * 16 + (lane >> 4) * 8 + r;
      out[((size_t)b * T_ + mq) * D_ + nn] = acc[t][r];
    }
  }
}

// ---------------------------------------------------------------------------
// Host launcher. Workspace (f16): Qp[4096*6144] | Val[4096*6144] |
// Kr[4096*768] | P[16*2048*2048]  -> ~230 MB total.
// ---------------------------------------------------------------------------
extern "C" void kernel_launch(void* const* d_in, const int* in_sizes, int n_in,
                              void* d_out, int out_size, void* d_ws, size_t ws_size,
                              hipStream_t stream) {
  const float* x  = (const float*)d_in[0];
  const float* Mw = (const float*)d_in[1];
  const float* Vw = (const float*)d_in[2];
  const int* positions = (const int*)d_in[3];
  float* out = (float*)d_out;

  _Float16* Qp  = (_Float16*)d_ws;
  _Float16* Val = Qp  + (size_t)BTM_ * HD_;
  _Float16* Kr  = Val + (size_t)BTM_ * HD_;
  _Float16* P   = Kr  + (size_t)BTM_ * D_;

  const dim3 blk(128);

  // 1+2: projections (Q and value) with on-the-fly f32->f16
  proj_gemm_kernel<<<dim3(HD_ / 64, BTM_ / 64), blk, 0, stream>>>(x, Mw, Qp);
  proj_gemm_kernel<<<dim3(HD_ / 64, BTM_ / 64), blk, 0, stream>>>(x, Vw, Val);

  // 3: RoPE on Q (in place) and K (from x)
  {
    const int total = B_ * T_ * (H_ + 1) * (D_ / 2);
    rope_kernel<<<(total + 255) / 256, 256, 0, stream>>>(x, positions, Qp, Kr);
  }

  // 4: causal scores -> P buffer (f16)
  scores_kernel<<<dim3(T_ / 64, T_ / 64, B_ * H_), blk, 0, stream>>>(Qp, Kr, P);

  // 5: row softmax with causal length + zero tail
  softmax_kernel<<<B_ * H_ * T_, 256, 0, stream>>>(P);

  // 6: out = sum_h P @ Val
  out_gemm_kernel<<<dim3(D_ / 64, T_ / 64, B_), blk, 0, stream>>>(P, Val, out);
}
